// LongformerSelfAttention_8065948581913
// MI455X (gfx1250) — matrix operand order repaired and checked by reading, not compiled, not run on hardware
//
#include <hip/hip_runtime.h>
#include <math.h>

// -------- CDNA5 WMMA types (gfx1250, wave32) --------
typedef __attribute__((ext_vector_type(16))) _Float16 v16h;
typedef __attribute__((ext_vector_type(8)))  _Float16 h8;
typedef __attribute__((ext_vector_type(8)))  float    v8f;

union HFrag { v16h v; _Float16 h[16]; h8 g[2]; };

#define E_    768
#define H_    12
#define DH_   64
#define SEQ_  4096
#define B_    2
#define WIN_  256
#define NG_   64
#define NEG_  (-1.0e9f)

__device__ __forceinline__ h8 cvt8(const float4& a, const float4& b) {
  h8 r;
  r[0] = (_Float16)a.x; r[1] = (_Float16)a.y; r[2] = (_Float16)a.z; r[3] = (_Float16)a.w;
  r[4] = (_Float16)b.x; r[5] = (_Float16)b.y; r[6] = (_Float16)b.z; r[7] = (_Float16)b.w;
  return r;
}

// ================================================================
// Kernel 1: tiled GEMM  C_f16[m,n] = (A_f32[m,:] @ W_f32[:,n] + bias[n]) * scale
// Block: 256 thr = 8 waves, block tile 64(M) x 128(N), K-step 32.
// Each wave: 2x2 grid of 16x16 WMMA tiles (32x32 output).
// Vectorized staging: per thread 2x float4 (A) + 4x float4 (B) b128 loads.
// remap=1: A row m -> (m/64)*SEQ + (m%64)   (first-64-rows per batch, for QG)
// ================================================================
#define GM 64
#define GN 128
#define GK 32
#define LPAD 16   // pad rows to 48 halves (96B): keeps h8 loads 16B-aligned

__global__ __launch_bounds__(256) void proj_gemm(
    const float* __restrict__ A, const float* __restrict__ W,
    const float* __restrict__ bias, _Float16* __restrict__ C,
    int M, float scale, int remap)
{
  __shared__ _Float16 As[GM][GK + LPAD];   // [m][k]
  __shared__ _Float16 Bs[GN][GK + LPAD];   // transposed [n][k]
  const int tid  = threadIdx.x;
  const int lane = tid & 31, wv = tid >> 5, hf = lane >> 4, ln = lane & 15;
  const int m0 = blockIdx.y * GM, n0 = blockIdx.x * GN;
  const int wm = (wv >> 2) * 32, wn = (wv & 3) * 32;  // 2(M) x 4(N) waves

  // fixed per-thread staging coordinates
  const int ar = tid >> 2,  ac = (tid & 3) * 8;   // A: row 0..63, col {0,8,16,24}
  const int bk = tid >> 3,  bn = (tid & 7) * 16;  // B: k 0..31,  n {0..112}
  const int gm = m0 + ar;
  const int arow = remap ? ((gm >> 6) * SEQ_ + (gm & 63)) : gm;
  const float* __restrict__ Abase = A + (size_t)arow * E_ + ac;
  const float* __restrict__ Wbase = W + (size_t)bk * E_ + n0 + bn;

  v8f zero = {};
  v8f acc[2][2] = {{zero, zero}, {zero, zero}};

  for (int k0 = 0; k0 < E_; k0 += GK) {
    // issue all global b128 loads up front (6 outstanding), then convert/store
    const float4* ap = (const float4*)(Abase + k0);
    const float4* bp = (const float4*)(Wbase + (size_t)k0 * E_);
    float4 a0 = ap[0], a1 = ap[1];
    float4 b0 = bp[0], b1 = bp[1], b2 = bp[2], b3 = bp[3];
    if (k0 + GK < E_) {   // prefetch next K-step tiles (global_prefetch_b8)
      __builtin_prefetch(Abase + k0 + GK, 0, 1);
      __builtin_prefetch(Wbase + (size_t)(k0 + GK) * E_, 0, 1);
    }
    *(h8*)&As[ar][ac] = cvt8(a0, a1);        // one 16B LDS store
    h8 c0 = cvt8(b0, b1), c1 = cvt8(b2, b3); // transpose B via scalar stores
#pragma unroll
    for (int j = 0; j < 8; ++j) Bs[bn + j][bk]     = c0[j];
#pragma unroll
    for (int j = 0; j < 8; ++j) Bs[bn + 8 + j][bk] = c1[j];
    __syncthreads();

    HFrag af[2], bf[2];
#pragma unroll
    for (int i = 0; i < 2; ++i) {
      const int row = wm + i * 16 + ln;
      af[i].g[0] = *(const h8*)&As[row][hf * 8];        // K = hf*8 + 0..7
      af[i].g[1] = *(const h8*)&As[row][16 + hf * 8];   // K = 16 + hf*8 + 0..7
    }
#pragma unroll
    for (int i = 0; i < 2; ++i) {
      const int col = wn + i * 16 + ln;
      bf[i].g[0] = *(const h8*)&Bs[col][hf * 8];        // K = hf*8 + 0..7
      bf[i].g[1] = *(const h8*)&Bs[col][16 + hf * 8];   // K = 16 + hf*8 + 0..7
    }
#pragma unroll
    for (int i = 0; i < 2; ++i)
#pragma unroll
      for (int j = 0; j < 2; ++j)
        acc[i][j] = __builtin_amdgcn_wmma_f32_16x16x32_f16(
            false, af[i].v, false, bf[j].v, (short)0, acc[i][j], false, false);
    __syncthreads();
  }

  // epilogue: C layout M = hf*8 + r, N = ln
#pragma unroll
  for (int i = 0; i < 2; ++i)
#pragma unroll
    for (int j = 0; j < 2; ++j)
#pragma unroll
      for (int r = 0; r < 8; ++r) {
        const int m = m0 + wm + i * 16 + hf * 8 + r;
        const int n = n0 + wn + j * 16 + ln;
        if (m < M)
          C[(size_t)m * E_ + n] = (_Float16)((acc[i][j][r] + bias[n]) * scale);
      }
}

// ================================================================
// Kernel 2: banded + global-column attention.
// Block: 128 thr = 4 waves; one block per (b, head, 16-query tile).
// Score buffer in LDS: cols 0..63 = global keys, 64..591 = band union
// u = j - (q0-256), u in [0,528); cols 592..607 are zero padding.
// ================================================================
#define NCOL 608

__global__ __launch_bounds__(128) void band_attn(
    const _Float16* __restrict__ Q, const _Float16* __restrict__ K,
    const _Float16* __restrict__ V, const float* __restrict__ amask,
    float* __restrict__ out)
{
  __shared__ __align__(16) float S[16][NCOL];   // ~38.9 KB of 320KB WGP LDS
  const int tid  = threadIdx.x;
  const int lane = tid & 31, wv = tid >> 5, hf = lane >> 4, ln = lane & 15;
  const int bid = blockIdx.x;
  const int qt  = bid & (SEQ_ / 16 - 1);
  const int hh  = (bid >> 8) % H_;
  const int bb  = bid / ((SEQ_ / 16) * H_);
  const int q0  = qt * 16;
  const float* amb = amask + (size_t)bb * SEQ_;
  const _Float16* Kb = K + (size_t)bb * SEQ_ * E_ + hh * DH_;
  const _Float16* Vb = V + (size_t)bb * SEQ_ * E_ + hh * DH_;

  // Q A-fragments: two 16B vector loads per 32-wide K chunk
  HFrag qa[2];
#pragma unroll
  for (int kc = 0; kc < 2; ++kc) {
    const _Float16* qrow = Q + ((size_t)bb * SEQ_ + q0 + ln) * E_ + hh * DH_ + kc * 32;
    qa[kc].g[0] = *(const h8*)(qrow + hf * 8);
    qa[kc].g[1] = *(const h8*)(qrow + 16 + hf * 8);
  }

  // ---- Phase 1: scores via WMMA QK^T (tile index uniform per wave) ----
  for (int t = wv; t < 37; t += 4) {
    const bool isg  = (t < 4);
    const int kbase = isg ? t * 16 : (q0 - WIN_ + (t - 4) * 16);
    const int key   = kbase + ln;
    const int keyc  = ::min(::max(key, 0), SEQ_ - 1);
    const _Float16* krow = Kb + (size_t)keyc * E_;
    v8f acc = {};
#pragma unroll
    for (int kc = 0; kc < 2; ++kc) {
      HFrag kf;  // B 32x16: lane = key column, dh hf*8+0..7 and 16+hf*8+0..7
      kf.g[0] = *(const h8*)(krow + kc * 32 + hf * 8);
      kf.g[1] = *(const h8*)(krow + kc * 32 + 16 + hf * 8);
      acc = __builtin_amdgcn_wmma_f32_16x16x32_f16(
          false, qa[kc].v, false, kf.v, (short)0, acc, false, false);
    }
#pragma unroll
    for (int r = 0; r < 8; ++r) {
      const int m = hf * 8 + r;
      if (isg) {
        S[m][kbase + ln] = acc[r];               // sel scores: no mask
      } else {
        const int u = (t - 4) * 16 + ln;
        const int j = kbase + ln;
        const int c = u - m;                      // band offset 0..512
        const bool valid = (j >= 0) && (j < SEQ_) && (c >= 0) && (c <= 2 * WIN_);
        float val = NEG_;
        if (valid) val = acc[r] + ((amb[j] != 0.f) ? -10000.f : 0.f);
        S[m][64 + u] = val;
      }
    }
  }
  __syncthreads();

  // ---- Phase 2: row softmax over cols 0..591 (8 threads / row) ----
  {
    const int row = tid >> 3, sub = tid & 7;
    float mx = -1e30f;
    for (int c = sub; c < 592; c += 8) mx = fmaxf(mx, S[row][c]);
    for (int o = 4; o >= 1; o >>= 1) mx = fmaxf(mx, __shfl_xor(mx, o, 8));
    float den = 0.f;
    for (int c = sub; c < 592; c += 8) den += __expf(S[row][c] - mx);
    for (int o = 4; o >= 1; o >>= 1) den += __shfl_xor(den, o, 8);
    const float kill = (amb[q0 + row] < 0.f) ? 0.f : 1.f;   // kpm zeroing
    const float inv  = kill / den;
    for (int c = sub; c < 592; c += 8) S[row][c] = __expf(S[row][c] - mx) * inv;
    if (sub < 2)
#pragma unroll
      for (int j2 = 0; j2 < 8; ++j2) S[row][592 + sub * 8 + j2] = 0.f;
  }
  __syncthreads();

  // ---- Phase 3: O = P @ V via WMMA; each wave owns one 16-wide dh tile ----
  {
    v8f oacc = {};
    for (int chn = 0; chn < NCOL / 32; ++chn) {
      HFrag pa, vf;
      const int cb = chn * 32;
      const float4* ps0 = (const float4*)&S[ln][cb + hf * 8];
      const float4* ps1 = (const float4*)&S[ln][cb + 16 + hf * 8];
      float4 p0 = ps0[0], p1 = ps0[1], p2 = ps1[0], p3 = ps1[1];
      pa.g[0] = cvt8(p0, p1);
      pa.g[1] = cvt8(p2, p3);
#pragma unroll
      for (int j = 0; j < 16; ++j) {
        const int colk = cb + ((j < 8) ? (hf * 8 + j) : (16 + hf * 8 + (j - 8)));
        int key = (colk < 64) ? colk : (q0 - WIN_ + (colk - 64));
        key = ::min(::max(key, 0), SEQ_ - 1);   // P==0 where clamped
        vf.h[j] = Vb[(size_t)key * E_ + wv * 16 + ln];
      }
      oacc = __builtin_amdgcn_wmma_f32_16x16x32_f16(
          false, pa.v, false, vf.v, (short)0, oacc, false, false);
    }
#pragma unroll
    for (int r = 0; r < 8; ++r) {
      const int m = hf * 8 + r;
      out[((size_t)bb * SEQ_ + q0 + m) * E_ + hh * DH_ + wv * 16 + ln] = oacc[r];
    }
  }
}

// ================================================================
// Kernel 3: global-token full attention (rows 0..63 per batch).
// Two-pass streaming softmax over all 4096 keys, 128-key LDS chunks.
// Block: 128 thr; one block per (b, head, 16-query tile of the 64).
// ================================================================
__global__ __launch_bounds__(128) void global_attn(
    const _Float16* __restrict__ QG, const _Float16* __restrict__ KG,
    const _Float16* __restrict__ VG, const float* __restrict__ amask,
    float* __restrict__ out)
{
  __shared__ __align__(16) float S[16][128];
  __shared__ float rowm[16], rowinv[16];
  const int tid  = threadIdx.x;
  const int lane = tid & 31, wv = tid >> 5, hf = lane >> 4, ln = lane & 15;
  const int bid = blockIdx.x;
  const int qt  = bid & 3;
  const int hh  = (bid >> 2) % H_;
  const int bb  = bid / (4 * H_);
  const int q0  = qt * 16;
  const float* amb = amask + (size_t)bb * SEQ_;
  const _Float16* Kb = KG + (size_t)bb * SEQ_ * E_ + hh * DH_;
  const _Float16* Vb = VG + (size_t)bb * SEQ_ * E_ + hh * DH_;

  HFrag qa[2];
#pragma unroll
  for (int kc = 0; kc < 2; ++kc) {
    const _Float16* qrow = QG + ((size_t)bb * NG_ + q0 + ln) * E_ + hh * DH_ + kc * 32;
    qa[kc].g[0] = *(const h8*)(qrow + hf * 8);
    qa[kc].g[1] = *(const h8*)(qrow + 16 + hf * 8);
  }

  const int row = tid >> 3, sub = tid & 7;
  float m_run = -1e30f, d_run = 0.f;

  // ---- Pass A: streaming row max / sum-exp ----
  for (int ch = 0; ch < SEQ_ / 128; ++ch) {
    const int k0 = ch * 128;
#pragma unroll
    for (int tt = 0; tt < 2; ++tt) {
      const int kb = k0 + wv * 32 + tt * 16;
      const _Float16* krow = Kb + (size_t)(kb + ln) * E_;
      __builtin_prefetch(krow + (size_t)128 * E_, 0, 1);  // next chunk
      v8f acc = {};
#pragma unroll
      for (int kc = 0; kc < 2; ++kc) {
        HFrag kf;
        kf.g[0] = *(const h8*)(krow + kc * 32 + hf * 8);
        kf.g[1] = *(const h8*)(krow + kc * 32 + 16 + hf * 8);
        acc = __builtin_amdgcn_wmma_f32_16x16x32_f16(
            false, qa[kc].v, false, kf.v, (short)0, acc, false, false);
      }
#pragma unroll
      for (int r = 0; r < 8; ++r) {
        const int m = hf * 8 + r;
        S[m][wv * 32 + tt * 16 + ln] =
            acc[r] + ((amb[kb + ln] < 0.f) ? NEG_ : 0.f);   // kpm mask
      }
    }
    __syncthreads();
    float mx = -1e30f;
    for (int c = sub; c < 128; c += 8) mx = fmaxf(mx, S[row][c]);
    for (int o = 4; o >= 1; o >>= 1) mx = fmaxf(mx, __shfl_xor(mx, o, 8));
    float dc = 0.f;
    for (int c = sub; c < 128; c += 8) dc += __expf(S[row][c] - mx);
    for (int o = 4; o >= 1; o >>= 1) dc += __shfl_xor(dc, o, 8);
    const float nm = fmaxf(m_run, mx);
    d_run = d_run * __expf(m_run - nm) + dc * __expf(mx - nm);
    m_run = nm;
    __syncthreads();
  }
  if (sub == 0) { rowm[row] = m_run; rowinv[row] = 1.f / d_run; }
  __syncthreads();

  // ---- Pass B: recompute scores, normalize, accumulate PV ----
  v8f oacc = {};
  for (int ch = 0; ch < SEQ_ / 128; ++ch) {
    const int k0 = ch * 128;
#pragma unroll
    for (int tt = 0; tt < 2; ++tt) {
      const int kb = k0 + wv * 32 + tt * 16;
      const _Float16* krow = Kb + (size_t)(kb + ln) * E_;
      v8f acc = {};
#pragma unroll
      for (int kc = 0; kc < 2; ++kc) {
        HFrag kf;
        kf.g[0] = *(const h8*)(krow + kc * 32 + hf * 8);
        kf.g[1] = *(const h8*)(krow + kc * 32 + 16 + hf * 8);
        acc = __builtin_amdgcn_wmma_f32_16x16x32_f16(
            false, qa[kc].v, false, kf.v, (short)0, acc, false, false);
      }
#pragma unroll
      for (int r = 0; r < 8; ++r) {
        const int m = hf * 8 + r;
        S[m][wv * 32 + tt * 16 + ln] =
            acc[r] + ((amb[kb + ln] < 0.f) ? NEG_ : 0.f);
      }
    }
    __syncthreads();
    for (int i = tid; i < 16 * 128; i += 128) {
      const int m = i >> 7, c = i & 127;
      S[m][c] = __expf(S[m][c] - rowm[m]) * rowinv[m];
    }
    __syncthreads();
#pragma unroll
    for (int sc2 = 0; sc2 < 4; ++sc2) {
      HFrag pa, vf;
      const int cb = sc2 * 32;
      const float4* ps0 = (const float4*)&S[ln][cb + hf * 8];
      const float4* ps1 = (const float4*)&S[ln][cb + 16 + hf * 8];
      float4 p0 = ps0[0], p1 = ps0[1], p2 = ps1[0], p3 = ps1[1];
      pa.g[0] = cvt8(p0, p1);
      pa.g[1] = cvt8(p2, p3);
#pragma unroll
      for (int j = 0; j < 16; ++j) {
        const int key = k0 + cb + ((j < 8) ? (hf * 8 + j) : (16 + hf * 8 + (j - 8)));
        vf.h[j] = Vb[(size_t)key * E_ + wv * 16 + ln];
      }
      oacc = __builtin_amdgcn_wmma_f32_16x16x32_f16(
          false, pa.v, false, vf.v, (short)0, oacc, false, false);
    }
    __syncthreads();
  }
#pragma unroll
  for (int r = 0; r < 8; ++r) {
    const int m = hf * 8 + r;
    out[((size_t)bb * SEQ_ + q0 + m) * E_ + hh * DH_ + wv * 16 + ln] = oacc[r];
  }
}

// ================================================================
// Host launcher
// ================================================================
extern "C" void kernel_launch(void* const* d_in, const int* in_sizes, int n_in,
                              void* d_out, int out_size, void* d_ws, size_t ws_size,
                              hipStream_t stream)
{
  (void)in_sizes; (void)n_in; (void)out_size; (void)ws_size;
  const float* hs  = (const float*)d_in[0];
  const float* am  = (const float*)d_in[1];
  const float* Wq  = (const float*)d_in[2];  const float* bq  = (const float*)d_in[3];
  const float* Wk  = (const float*)d_in[4];  const float* bk  = (const float*)d_in[5];
  const float* Wv  = (const float*)d_in[6];  const float* bv  = (const float*)d_in[7];
  const float* Wqg = (const float*)d_in[8];  const float* bqg = (const float*)d_in[9];
  const float* Wkg = (const float*)d_in[10]; const float* bkg = (const float*)d_in[11];
  const float* Wvg = (const float*)d_in[12]; const float* bvg = (const float*)d_in[13];
  float* out = (float*)d_out;

  const int M = B_ * SEQ_;                 // 8192
  const size_t PROJ = (size_t)M * E_;      // elems per projection
  _Float16* Q  = (_Float16*)d_ws;          // ~63 MB total f16 workspace
  _Float16* K  = Q  + PROJ;
  _Float16* V  = K  + PROJ;
  _Float16* KG = V  + PROJ;
  _Float16* VG = KG + PROJ;
  _Float16* QG = VG + PROJ;                // [b, 64, E]
  const float scale = 0.125f;              // 1/sqrt(64)

  dim3 blk(256);
  dim3 gmain(E_ / GN, M / GM);             // (6, 128)
  proj_gemm<<<gmain, blk, 0, stream>>>(hs, Wq,  bq,  Q,  M, scale, 0);
  proj_gemm<<<gmain, blk, 0, stream>>>(hs, Wk,  bk,  K,  M, 1.f,   0);
  proj_gemm<<<gmain, blk, 0, stream>>>(hs, Wv,  bv,  V,  M, 1.f,   0);
  proj_gemm<<<gmain, blk, 0, stream>>>(hs, Wkg, bkg, KG, M, 1.f,   0);
  proj_gemm<<<gmain, blk, 0, stream>>>(hs, Wvg, bvg, VG, M, 1.f,   0);
  dim3 gqg(E_ / GN, (B_ * NG_) / GM);      // (6, 2)
  proj_gemm<<<gqg, blk, 0, stream>>>(hs, Wqg, bqg, QG, B_ * NG_, scale, 1);

  band_attn<<<B_ * H_ * (SEQ_ / 16), 128, 0, stream>>>(Q, K, V, am, out);
  global_attn<<<B_ * H_ * (NG_ / 16), 128, 0, stream>>>(QG, KG, VG, am, out);
}
